// ParallelNCA_58162447123096
// MI455X (gfx1250) — matrix operand order repaired and checked by reading, not compiled, not run on hardware
//
#include <hip/hip_runtime.h>

typedef __attribute__((ext_vector_type(16))) __bf16 v16bf;
typedef __attribute__((ext_vector_type(8)))  float  v8f;

#define NM   64
#define CIN  16
#define HID  128
#define KDIM 144   // 9*16
#define HH   32
#define WW   32

union V16 {
    uint4 q[2];
    v16bf v;
};

__global__ __launch_bounds__(128) void nca_wmma_kernel(
    const float* __restrict__ x,  const float* __restrict__ w1,
    const float* __restrict__ b1, const float* __restrict__ w2,
    const float* __restrict__ b2, float* __restrict__ out)
{
    __shared__ __bf16 w1S[HID * KDIM];     // 36864 B, [d][k] = B-matrix [N][K]
    __shared__ __bf16 w2S[CIN * HID];      //  4096 B, [c][d] = B-matrix [N][K]
    __shared__ float  b1S[HID];
    __shared__ float  b2S[CIN];
    __shared__ __bf16 percS[4][16 * KDIM]; // per-wave A tile [16 pix][144]; reused as hidden [16][128]

    const int t    = threadIdx.x;
    const int lane = t & 31;
    const int wid  = t >> 5;
    const int rl   = lane & 15;   // row/col-within-tile index
    const int kh   = lane >> 4;   // K-half selector for WMMA operand layouts

    const int blk = blockIdx.x;
    const int g   = blk >> 8;          // 64 groups
    const int sp  = blk & 255;
    const int b   = sp >> 4;           // batch
    const int h0  = (sp & 15) << 1;    // 2 rows per WG
    const int h   = h0 + (wid >> 1);
    const int w0  = (wid & 1) << 4;    // 16-pixel tile along W

    // ---------------- stage weights / biases to LDS (f32 -> bf16) ----------------
    const float* w1g = w1 + (long)g * (HID * KDIM);
    for (int e = t; e < HID * KDIM; e += 128) w1S[e] = (__bf16)w1g[e];
    const float* w2g = w2 + (long)g * (CIN * HID);
    for (int e = t; e < CIN * HID; e += 128) w2S[e] = (__bf16)w2g[e];
    if (t < HID) b1S[t] = b1[g * HID + t];
    if (t < CIN) b2S[t] = b2[g * CIN + t];

    // ---------------- build perception A-tile: 16 pixels x 144 (bf16) -------------
    // Reference grouping: flat perception channel i = n*1024 + mc, groups are
    // contiguous 144-slices: i = g*144 + k -> n = i>>10 (shift), mc = i & 1023.
    __bf16* perc = &percS[wid][0];
    for (int tt = 0; tt < 72; ++tt) {
        int e = tt * 32 + lane;      // covers 16*144 elements across 32 lanes
        int r = e & 15;              // pixel within tile
        int k = e >> 4;              // K index 0..143
        int i = g * KDIM + k;
        int n = i >> 10;
        int mc = i & 1023;
        int rr, cc;
        if (n == 0) { rr = 0; cc = 0; }
        else { int jj = (n - 1) + (((n - 1) >= 4) ? 1 : 0); rr = jj / 3 - 1; cc = jj % 3 - 1; }
        int hh = h + rr;
        int ww = w0 + r + cc;
        float val = 0.0f;
        if ((unsigned)hh < HH && (unsigned)ww < WW)
            val = x[((long)((b << 10) + mc)) * (HH * WW) + hh * WW + ww];
        perc[r * KDIM + k] = (__bf16)val;
    }
    __syncthreads();

    // ---------------- load A operand registers (5 K-steps of 32, K padded) --------
    // A 16x32 k order chosen here (not ISA table order): lane rl = row M; K[kh*16 .. +15] contiguous (matches B)
    V16 a1[5];
    {
        const __bf16* rowp = perc + rl * KDIM;
        #pragma unroll
        for (int s = 0; s < 5; ++s) {
            int k0 = s * 32 + kh * 16;
            int k1 = k0 + 8;
            if (k1 < KDIM) {
                a1[s].q[0] = *reinterpret_cast<const uint4*>(rowp + k0);
                a1[s].q[1] = *reinterpret_cast<const uint4*>(rowp + k1);
            } else {
                a1[s].q[0] = make_uint4(0u, 0u, 0u, 0u);
                a1[s].q[1] = make_uint4(0u, 0u, 0u, 0u);
            }
        }
    }

    // ---------------- matmul1: hidden[16x128] = ReLU(A[16x144] * w1^T + b1) -------
    __bf16* hid = perc;  // reuse wave LDS region as hidden [16][HID] bf16
    #pragma unroll 2
    for (int dt = 0; dt < 8; ++dt) {
        v8f c = {0.f, 0.f, 0.f, 0.f, 0.f, 0.f, 0.f, 0.f};
        const __bf16* bcol = w1S + (dt * 16 + rl) * KDIM;  // N = rl, contiguous K
        #pragma unroll
        for (int s = 0; s < 5; ++s) {
            int kb = s * 32 + kh * 16;   // B k order chosen here, same as A: lane rl = col N, K[kh*16 .. +15]
            V16 bv;
            if (kb < KDIM) {
                bv.q[0] = *reinterpret_cast<const uint4*>(bcol + kb);
                bv.q[1] = *reinterpret_cast<const uint4*>(bcol + kb + 8);
            } else {
                bv.q[0] = make_uint4(0u, 0u, 0u, 0u);
                bv.q[1] = make_uint4(0u, 0u, 0u, 0u);
            }
            c = __builtin_amdgcn_wmma_f32_16x16x32_bf16(
                    false, a1[s].v, false, bv.v, (short)0, c, false, false);
        }
        // D layout: VGPR v -> row M = v + kh*8, col N = rl
        float bias = b1S[dt * 16 + rl];
        __bf16* hcol = hid + dt * 16 + rl;
        #pragma unroll
        for (int v = 0; v < 8; ++v) {
            float hval = fmaxf(c[v] + bias, 0.0f);
            hcol[(v + kh * 8) * HID] = (__bf16)hval;
        }
    }
    // DS ops from the same wave complete in order; compiler inserts s_wait_dscnt
    // for the RAW on the hidden tile below.

    // ---------------- matmul2: dx[16x16] = hidden[16x128] * w2^T ------------------
    v8f c2 = {0.f, 0.f, 0.f, 0.f, 0.f, 0.f, 0.f, 0.f};
    {
        const __bf16* hrow  = hid + rl * HID;   // A: M = rl
        const __bf16* b2col = w2S + rl * HID;   // B: N = rl (c_out)
        #pragma unroll
        for (int s = 0; s < 4; ++s) {
            V16 av, bv;
            int ka = s * 32 + kh * 16;
            av.q[0] = *reinterpret_cast<const uint4*>(hrow + ka);
            av.q[1] = *reinterpret_cast<const uint4*>(hrow + ka + 8);
            int kb = s * 32 + kh * 16;
            bv.q[0] = *reinterpret_cast<const uint4*>(b2col + kb);
            bv.q[1] = *reinterpret_cast<const uint4*>(b2col + kb + 8);
            c2 = __builtin_amdgcn_wmma_f32_16x16x32_bf16(
                     false, av.v, false, bv.v, (short)0, c2, false, false);
        }
    }

    // ---------------- epilogue: out = x + dx + b2 ---------------------------------
    // Lane rl = output channel within group; c2[v] -> pixel w = w0 + kh*8 + v.
    {
        const int  ch   = g * CIN + rl;
        const long base = ((long)((b << 10) + ch)) * (HH * WW) + h * WW + w0 + kh * 8;
        const float bias2 = b2S[rl];
        const float4 x0 = *reinterpret_cast<const float4*>(x + base);
        const float4 x1 = *reinterpret_cast<const float4*>(x + base + 4);
        float4 o0, o1;
        o0.x = x0.x + c2[0] + bias2;  o0.y = x0.y + c2[1] + bias2;
        o0.z = x0.z + c2[2] + bias2;  o0.w = x0.w + c2[3] + bias2;
        o1.x = x1.x + c2[4] + bias2;  o1.y = x1.y + c2[5] + bias2;
        o1.z = x1.z + c2[6] + bias2;  o1.w = x1.w + c2[7] + bias2;
        *reinterpret_cast<float4*>(out + base)     = o0;
        *reinterpret_cast<float4*>(out + base + 4) = o1;
    }
}

extern "C" void kernel_launch(void* const* d_in, const int* in_sizes, int n_in,
                              void* d_out, int out_size, void* d_ws, size_t ws_size,
                              hipStream_t stream) {
    (void)in_sizes; (void)n_in; (void)d_ws; (void)ws_size; (void)out_size;
    const float* x  = (const float*)d_in[0];
    const float* w1 = (const float*)d_in[1];
    const float* b1 = (const float*)d_in[2];
    const float* w2 = (const float*)d_in[3];
    const float* b2 = (const float*)d_in[4];
    float* out = (float*)d_out;

    // 64 groups x 16 batches x 16 row-pairs = 16384 workgroups, 128 threads (4 waves)
    dim3 grid(NM * 256);
    nca_wmma_kernel<<<grid, 128, 0, stream>>>(x, w1, b1, w2, b2, out);
}
